// SentinelMamba_47811575939615
// MI455X (gfx1250) — matrix operand order repaired and checked by reading, not compiled, not run on hardware
//
#include <hip/hip_runtime.h>
#include <cstdint>
#include <cstddef>

// ---------------- problem constants (from reference) ----------------
#define D_MODEL 1024
#define D_INNER 2048     // EXPAND * D_MODEL
#define DT_RANK 64       // ceil(1024/16)
#define D_STATE 16
#define BATCH   2
#define LEN     1024
#define NTOK    (BATCH*LEN)      // 2048 tokens
#define NCHUNK  8
#define CHUNK   (LEN/NCHUNK)     // 128
#define NCLS    10

typedef __attribute__((ext_vector_type(16))) __bf16         bf16x16;
typedef __attribute__((ext_vector_type(8)))  float          f32x8;
typedef __attribute__((ext_vector_type(8)))  unsigned short u16x8;
typedef __attribute__((ext_vector_type(4)))  int            v4i;

union FragAB { bf16x16 v; u16x8 h[2]; };

// CDNA5 async global->LDS copies (ASYNCcnt-tracked), if this toolchain has them
#if defined(__has_builtin)
#if __has_builtin(__builtin_amdgcn_global_load_async_to_lds_b128) && \
    __has_builtin(__builtin_amdgcn_s_wait_asynccnt)
#define HAS_ASYNC_LDS 1
#endif
#endif
#ifndef HAS_ASYNC_LDS
#define HAS_ASYNC_LDS 0
#endif

#define ASG __attribute__((address_space(1)))
#define ASL __attribute__((address_space(3)))

// float -> bf16 (round to nearest even), pure bit manipulation
__device__ __forceinline__ unsigned short f2bf(float f) {
  union { float f; unsigned int u; } c; c.f = f;
  unsigned int u = c.u;
  unsigned int r = u + 0x7FFFu + ((u >> 16) & 1u);
  if ((u & 0x7F800000u) == 0x7F800000u) r = u;   // inf/nan passthrough
  return (unsigned short)(r >> 16);
}

__global__ void k_cvt_bf16(const float* __restrict__ src,
                           unsigned short* __restrict__ dst, int n) {
  int i = blockIdx.x * 256 + threadIdx.x;
  if (i < n) dst[i] = f2bf(src[i]);
}

// stage one 128x32 A-tile + 128x32 B-tile into LDS.
// Out-of-range rows are CLAMPED (not zero-filled): a WMMA output column/row
// depends only on its own B column / A row, and OOB outputs are never stored,
// so clamping is safe and keeps the staging loads unpredicated.
__device__ __forceinline__ void stage_tile(
    const unsigned short* __restrict__ A, int lda, int M,
    const unsigned short* __restrict__ Bw, int ldb, int N,
    unsigned short* As, unsigned short* Bs,
    int m0, int n0, int kb, int t) {
#pragma unroll
  for (int it = 0; it < 2; it++) {
    int q   = t + it * 256;        // 0..511
    int row = q >> 2;
    int kc  = (q & 3) << 3;        // 0,8,16,24
    int gm = m0 + row; if (gm >= M) gm = M - 1;
    int gn = n0 + row; if (gn >= N) gn = N - 1;
    const unsigned short* ga = A  + (size_t)gm * lda + kb + kc;
    const unsigned short* gb = Bw + (size_t)gn * ldb + kb + kc;
    unsigned short* la = As + row * 40 + kc;
    unsigned short* lb = Bs + row * 40 + kc;
#if HAS_ASYNC_LDS
    __builtin_amdgcn_global_load_async_to_lds_b128(
        (ASG v4i*)ga, (ASL v4i*)la, 0, 0);
    __builtin_amdgcn_global_load_async_to_lds_b128(
        (ASG v4i*)gb, (ASL v4i*)lb, 0, 0);
#else
    *(u16x8*)la = *(const u16x8*)ga;
    *(u16x8*)lb = *(const u16x8*)gb;
#endif
  }
}

// ------------------------------------------------------------------
// Generic bf16 WMMA GEMM:  C[M][N] = A[M][K] * Bw[N][K]^T  (fp32 acc)
// Block: 256 threads = 8 waves, tile 128(M) x 128(N), K-step 32,
// double-buffered LDS with async global->LDS staging.
// Wave (wm in 0..3, wn in 0..1) computes 32(M) x 64(N) = 2x4 WMMA tiles.
// op==1: epilogue softplus(acc + bias[n])  (for delta)
// ------------------------------------------------------------------
__launch_bounds__(256)
__global__ void k_gemm_bf16(const unsigned short* __restrict__ A, int lda,
                            const unsigned short* __restrict__ Bw, int ldb,
                            float* __restrict__ C, int ldc,
                            int M, int N, int K,
                            int op, const float* __restrict__ bias) {
  // pitch 40 shorts (80B) keeps 16B alignment and staggers banks
  __shared__ unsigned short As[2][128 * 40];
  __shared__ unsigned short Bs[2][128 * 40];

  const int t    = threadIdx.x;
  const int m0   = blockIdx.y * 128;
  const int n0   = blockIdx.x * 128;
  const int w    = t >> 5;
  const int lane = t & 31;
  const int lm   = lane & 15;      // column/row within 16-wide tile
  const int hi   = lane >> 4;      // lane half selects K sub-range
  const int wm   = w >> 1;         // 0..3
  const int wn   = w & 1;          // 0..1

  f32x8 acc[2][4];
#pragma unroll
  for (int i = 0; i < 2; i++)
#pragma unroll
    for (int j = 0; j < 4; j++) acc[i][j] = (f32x8){0,0,0,0,0,0,0,0};

  // prologue: stage first K-tile into buffer 0
  stage_tile(A, lda, M, Bw, ldb, N, As[0], Bs[0], m0, n0, 0, t);

  int nbuf = 0;
  for (int kb = 0; kb < K; kb += 32) {
    const int cur = nbuf;
    nbuf ^= 1;
    const bool has_next = (kb + 32) < K;
    if (has_next)
      stage_tile(A, lda, M, Bw, ldb, N, As[nbuf], Bs[nbuf], m0, n0, kb + 32, t);
#if HAS_ASYNC_LDS
    // in-order completion: waiting to <=4 outstanding guarantees the current
    // stage's 4 copies have landed in LDS
    if (has_next) __builtin_amdgcn_s_wait_asynccnt(4);
    else          __builtin_amdgcn_s_wait_asynccnt(0);
#endif
    __syncthreads();   // current tile visible to all waves

    // ---- build per-lane fragments per ISA 7.12.2 layouts
    // A 16x32 bf16: lane m=lm; half hi: j0..7 -> K=hi*8+j, j8..15 -> K=hi*8+16+j
    FragAB af[2];
#pragma unroll
    for (int i = 0; i < 2; i++) {
      const unsigned short* p = &As[cur][(wm * 32 + i * 16 + lm) * 40 + hi * 8];
      af[i].h[0] = *(const u16x8*)p;
      af[i].h[1] = *(const u16x8*)(p + 16);
    }
    // B 32x16 bf16: lane n=lm; half hi: j0..7 -> K=hi*8+j, j8..15 -> K=hi*8+16+j
    FragAB bfm[4];
#pragma unroll
    for (int j = 0; j < 4; j++) {
      const unsigned short* p = &Bs[cur][(wn * 64 + j * 16 + lm) * 40 + hi * 8];
      bfm[j].h[0] = *(const u16x8*)p;
      bfm[j].h[1] = *(const u16x8*)(p + 16);
    }

#pragma unroll
    for (int i = 0; i < 2; i++)
#pragma unroll
      for (int j = 0; j < 4; j++)
        acc[i][j] = __builtin_amdgcn_wmma_f32_16x16x32_bf16(
            false, af[i].v, false, bfm[j].v, (short)0, acc[i][j], false, false);

    __syncthreads();   // all reads of buffer `cur` done before it is restaged
  }

  // ---- epilogue: D layout — VGPR r on lane: M = r + 8*hi, N = lm
#pragma unroll
  for (int i = 0; i < 2; i++) {
    int mb = m0 + wm * 32 + i * 16 + hi * 8;
#pragma unroll
    for (int j = 0; j < 4; j++) {
      int gn = n0 + wn * 64 + j * 16 + lm;
      if (gn < N) {
        float bv = (op == 1) ? bias[gn] : 0.f;
#pragma unroll
        for (int r = 0; r < 8; r++) {
          int gm = mb + r;
          if (gm < M) {
            float v = acc[i][j][r];
            if (op == 1) { v += bv; v = (v > 20.f) ? v : log1pf(__expf(v)); }
            C[(size_t)gm * ldc + gn] = v;
          }
        }
      }
    }
  }
}

// causal depthwise conv (D_CONV=4) + bias + SiLU : xp -> u
__global__ void k_conv_silu(const float* __restrict__ xp,
                            const float* __restrict__ Wc,
                            const float* __restrict__ bc,
                            float* __restrict__ u) {
  int t = blockIdx.x * 256 + threadIdx.x;
  if (t >= NTOK * D_INNER) return;
  int d   = t % D_INNER;
  int tok = t / D_INNER;
  int l   = tok & (LEN - 1);
  float acc = bc[d];
#pragma unroll
  for (int j = 0; j < 4; j++) {
    int li = l - 3 + j;
    if (li >= 0) acc += xp[(size_t)(tok - 3 + j) * D_INNER + d] * Wc[d * 4 + j];
  }
  u[t] = acc / (1.f + __expf(-acc));   // silu
}

// selective-scan chunk: per (b, chunk, d, s) compute (prod dA, folded dBu)
__global__ void k_scan_chunk(const float* __restrict__ delta,
                             const float* __restrict__ u,
                             const float* __restrict__ xdbl,
                             const float* __restrict__ A_log,
                             float* __restrict__ hA,
                             float* __restrict__ hB) {
  int t = blockIdx.x * 256 + threadIdx.x;        // 0 .. 2*8*2048*16-1
  int s = t & 15;
  int d = (t >> 4) & (D_INNER - 1);
  int c = (t >> 15) & 7;
  int b = t >> 18;
  float Aval = -__expf(A_log[d * D_STATE + s]);
  float ap = 1.f, bp = 0.f;
  int tok0 = b * LEN + c * CHUNK;
  for (int il = 0; il < CHUNK; ++il) {
    int tok  = tok0 + il;
    float dl = delta[(size_t)tok * D_INNER + d];
    float uv = u[(size_t)tok * D_INNER + d];
    float Bv = xdbl[tok * 96 + DT_RANK + s];
    float dA = __expf(dl * Aval);
    bp = dA * bp + dl * Bv * uv;
    ap *= dA;
  }
  hA[t] = ap;
  hB[t] = bp;
}

// combine chunks -> h[L-1]; y_last = h.C + D*u; yz = y * silu(z_last)
__global__ void k_finalize(const float* __restrict__ hA,
                           const float* __restrict__ hB,
                           const float* __restrict__ xdbl,
                           const float* __restrict__ u,
                           const float* __restrict__ Dp,
                           const float* __restrict__ z_last,
                           float* __restrict__ yz) {
  int t = blockIdx.x * 256 + threadIdx.x;
  if (t >= BATCH * D_INNER) return;
  int d = t % D_INNER, b = t / D_INNER;
  int tokL = b * LEN + LEN - 1;
  float y = 0.f;
#pragma unroll
  for (int s = 0; s < D_STATE; ++s) {
    float h = 0.f;
#pragma unroll
    for (int c = 0; c < NCHUNK; ++c) {
      size_t idx = ((size_t)((b * NCHUNK + c) * D_INNER + d)) * D_STATE + s;
      h = hA[idx] * h + hB[idx];
    }
    y += h * xdbl[tokL * 96 + DT_RANK + D_STATE + s];   // Cs
  }
  y += Dp[d] * u[(size_t)tokL * D_INNER + d];
  float z = z_last[b * D_INNER + d];
  yz[t] = y * (z / (1.f + __expf(-z)));
}

// fp32 GEMV: out[b][n] = sum_k vin[b*pv + k] * W[n*ldw + k] (+bias); wave per output
__global__ void k_gemv(const float* __restrict__ vin, int pv,
                       const float* __restrict__ W, int ldw,
                       const float* __restrict__ bias,
                       float* __restrict__ out, int po,
                       int Bc, int N, int K) {
  int gw   = blockIdx.x * 8 + (threadIdx.x >> 5);
  int lane = threadIdx.x & 31;
  if (gw >= Bc * N) return;
  int b = gw / N, n = gw % N;
  const float* vp = vin + (size_t)b * pv;
  const float* wp = W + (size_t)n * ldw;
  float s = 0.f;
  for (int k = lane; k < K; k += 32) s += vp[k] * wp[k];
#pragma unroll
  for (int off = 16; off > 0; off >>= 1) s += __shfl_xor(s, off, 32);
  if (lane == 0) out[(size_t)b * po + n] = s + (bias ? bias[n] : 0.f);
}

// ------------------------------------------------------------------
extern "C" void kernel_launch(void* const* d_in, const int* in_sizes, int n_in,
                              void* d_out, int out_size, void* d_ws, size_t ws_size,
                              hipStream_t stream) {
  const float* x      = (const float*)d_in[0];   // (2,1024,1024)
  const float* W_in   = (const float*)d_in[1];   // (4096,1024)
  const float* W_conv = (const float*)d_in[2];   // (2048,1,4)
  const float* b_conv = (const float*)d_in[3];   // (2048)
  const float* W_xprj = (const float*)d_in[4];   // (96,2048)
  const float* W_dt   = (const float*)d_in[5];   // (2048,64)
  const float* b_dt   = (const float*)d_in[6];   // (2048)
  const float* A_log  = (const float*)d_in[7];   // (2048,16)
  const float* Dp     = (const float*)d_in[8];   // (2048)
  const float* W_out  = (const float*)d_in[9];   // (1024,2048)
  const float* W_fc   = (const float*)d_in[10];  // (10,1024)
  const float* b_fc   = (const float*)d_in[11];  // (10)
  float* out = (float*)d_out;                    // (2,10)

  // workspace carve-out (256B aligned)
  size_t off = 0;
  auto alloc = [&](size_t bytes) -> void* {
    void* p = (char*)d_ws + off;
    off = (off + bytes + 255) & ~(size_t)255;
    return p;
  };
  float* xp_f    = (float*)alloc((size_t)NTOK * D_INNER * 4);
  float* u_f     = (float*)alloc((size_t)NTOK * D_INNER * 4);
  float* delta_f = (float*)alloc((size_t)NTOK * D_INNER * 4);
  float* xdbl_f  = (float*)alloc((size_t)NTOK * 96 * 4);
  unsigned short* x_bf    = (unsigned short*)alloc((size_t)NTOK * D_MODEL * 2);
  unsigned short* win_bf  = (unsigned short*)alloc((size_t)D_INNER * D_MODEL * 2);
  unsigned short* u_bf    = (unsigned short*)alloc((size_t)NTOK * D_INNER * 2);
  unsigned short* wxp_bf  = (unsigned short*)alloc((size_t)96 * D_INNER * 2);
  unsigned short* wdt_bf  = (unsigned short*)alloc((size_t)D_INNER * DT_RANK * 2);
  unsigned short* xdbl_bf = (unsigned short*)alloc((size_t)NTOK * 96 * 2);
  float* z_last  = (float*)alloc((size_t)BATCH * D_INNER * 4);
  float* hA      = (float*)alloc((size_t)BATCH * NCHUNK * D_INNER * D_STATE * 4);
  float* hB      = (float*)alloc((size_t)BATCH * NCHUNK * D_INNER * D_STATE * 4);
  float* yz      = (float*)alloc((size_t)BATCH * D_INNER * 4);
  float* olast   = (float*)alloc((size_t)BATCH * D_MODEL * 4);

  // 1) bf16 conversions of GEMM operands
  int n;
  n = NTOK * D_MODEL;
  k_cvt_bf16<<<(n + 255) / 256, 256, 0, stream>>>(x, x_bf, n);
  n = D_INNER * D_MODEL;                       // first 2048 rows of W_in (xp part)
  k_cvt_bf16<<<(n + 255) / 256, 256, 0, stream>>>(W_in, win_bf, n);
  n = 96 * D_INNER;
  k_cvt_bf16<<<(n + 255) / 256, 256, 0, stream>>>(W_xprj, wxp_bf, n);
  n = D_INNER * DT_RANK;
  k_cvt_bf16<<<(n + 255) / 256, 256, 0, stream>>>(W_dt, wdt_bf, n);

  // 2) xp = x @ W_in[:2048].T   (M=2048, N=2048, K=1024)
  k_gemm_bf16<<<dim3(D_INNER / 128, NTOK / 128), 256, 0, stream>>>(
      x_bf, D_MODEL, win_bf, D_MODEL, xp_f, D_INNER,
      NTOK, D_INNER, D_MODEL, 0, nullptr);

  // 3) z at last token only (fp32 GEMV): z_last[b][n] = x[b,L-1,:] . W_in[2048+n,:]
  k_gemv<<<(BATCH * D_INNER + 7) / 8, 256, 0, stream>>>(
      x + (size_t)(LEN - 1) * D_MODEL, LEN * D_MODEL,
      W_in + (size_t)D_INNER * D_MODEL, D_MODEL,
      nullptr, z_last, D_INNER, BATCH, D_INNER, D_MODEL);

  // 4) depthwise conv + SiLU -> u
  n = NTOK * D_INNER;
  k_conv_silu<<<(n + 255) / 256, 256, 0, stream>>>(xp_f, W_conv, b_conv, u_f);

  // 5) u -> bf16 ; x_dbl = u @ W_xproj.T  (M=2048, N=96, K=2048)
  k_cvt_bf16<<<(n + 255) / 256, 256, 0, stream>>>(u_f, u_bf, n);
  k_gemm_bf16<<<dim3(1, NTOK / 128), 256, 0, stream>>>(
      u_bf, D_INNER, wxp_bf, D_INNER, xdbl_f, 96,
      NTOK, 96, D_INNER, 0, nullptr);

  // 6) delta = softplus(dt @ W_dt.T + b_dt)  (dt = x_dbl[:, :64], lda=96)
  n = NTOK * 96;
  k_cvt_bf16<<<(n + 255) / 256, 256, 0, stream>>>(xdbl_f, xdbl_bf, n);
  k_gemm_bf16<<<dim3(D_INNER / 128, NTOK / 128), 256, 0, stream>>>(
      xdbl_bf, 96, wdt_bf, DT_RANK, delta_f, D_INNER,
      NTOK, D_INNER, DT_RANK, 1, b_dt);

  // 7) chunked selective scan (exact linear-recurrence composition)
  n = BATCH * NCHUNK * D_INNER * D_STATE;      // 524288
  k_scan_chunk<<<n / 256, 256, 0, stream>>>(delta_f, u_f, xdbl_f, A_log, hA, hB);

  // 8) combine chunks + y_last + gating
  k_finalize<<<(BATCH * D_INNER + 255) / 256, 256, 0, stream>>>(
      hA, hB, xdbl_f, u_f, Dp, z_last, yz);

  // 9) out_last = yz @ W_out.T  (2 x 1024, K=2048)
  k_gemv<<<(BATCH * D_MODEL + 7) / 8, 256, 0, stream>>>(
      yz, D_INNER, W_out, D_INNER, nullptr, olast, D_MODEL,
      BATCH, D_MODEL, D_INNER);

  // 10) logits = out_last @ W_fc.T + b_fc  (2 x 10)
  k_gemv<<<(BATCH * NCLS + 7) / 8, 256, 0, stream>>>(
      olast, D_MODEL, W_fc, D_MODEL, b_fc, out, NCLS,
      BATCH, NCLS, D_MODEL);
}
